// LinearAttention2D_71768903517051
// MI455X (gfx1250) — matrix operand order repaired and checked by reading, not compiled, not run on hardware
//
#include <hip/hip_runtime.h>
#include <hip/hip_bf16.h>
#include <math.h>

typedef __attribute__((ext_vector_type(16))) _Float16 v16h;
typedef __attribute__((ext_vector_type(8)))  _Float16 v8h;
typedef __attribute__((ext_vector_type(8)))  float    v8f;
typedef __attribute__((ext_vector_type(4)))  int      v4i;

#define CH   512
#define NHD  8
#define HD   64
#define BATCH 8
#define NTOK 4096              // 64*64 tokens per image
#define MROWS (BATCH*NTOK)     // 32768

// ---- gfx1250 async global->LDS copy (ASYNCcnt path), guarded by __has_builtin
#if defined(__has_builtin)
#  if __has_builtin(__builtin_amdgcn_global_load_async_to_lds_b128)
#    define HAVE_ASYNC_LDS 1
#  endif
#endif
#ifndef HAVE_ASYNC_LDS
#  define HAVE_ASYNC_LDS 0
#endif

#if HAVE_ASYNC_LDS
typedef __attribute__((address_space(1))) v4i gv4i_t;
typedef __attribute__((address_space(3))) v4i lv4i_t;
__device__ __forceinline__ void g2l_b128(const _Float16* g, _Float16* l) {
  __builtin_amdgcn_global_load_async_to_lds_b128(
      (gv4i_t*)(void*)g,
      (lv4i_t*)(void*)l, 0, 0);
}
__device__ __forceinline__ void async_wait() {
#  if __has_builtin(__builtin_amdgcn_s_wait_asynccnt)
  __builtin_amdgcn_s_wait_asynccnt(0);
#  else
  asm volatile("s_wait_asynccnt 0x0" ::: "memory");
#  endif
}
#else
__device__ __forceinline__ void g2l_b128(const _Float16* g, _Float16* l) {
  *(v8h*)l = *(const v8h*)g;
}
__device__ __forceinline__ void async_wait() {}
#endif

__device__ __forceinline__ v16h cat16(v8h a, v8h b) {
  return __builtin_shufflevector(a, b, 0,1,2,3,4,5,6,7,8,9,10,11,12,13,14,15);
}

// ---------------------------------------------------------------- zero init
__global__ void k_zero(float* __restrict__ p, int n) {
  int i = blockIdx.x * 256 + threadIdx.x;
  if (i < n) p[i] = 0.f;
}

// ---------------------------------------------------- x (f32) -> xh (f16)
__global__ void k_xconv(const float* __restrict__ x, _Float16* __restrict__ xh) {
  size_t i = ((size_t)blockIdx.x * 256 + threadIdx.x) * 8;
  const float4* p = (const float4*)(x + i);
  float4 f0 = p[0], f1 = p[1];
  v8h a = {(_Float16)f0.x, (_Float16)f0.y, (_Float16)f0.z, (_Float16)f0.w,
           (_Float16)f1.x, (_Float16)f1.y, (_Float16)f1.z, (_Float16)f1.w};
  *(v8h*)(xh + i) = a;
}

// ------------------------------------------- W (f32 [K][N]) -> Wt (f16 [N][K])
__global__ void k_wconv(const float* __restrict__ W, _Float16* __restrict__ Wt) {
  int i = blockIdx.x * 256 + threadIdx.x;      // 0 .. 262143
  int n = i >> 9, kk = i & 511;
  Wt[(size_t)n * CH + kk] = (_Float16)W[(size_t)kk * CH + n];
}

// ---------------------------------------------------------------- main GEMM
// D[m,n] = act( sum_k A[m,k] * Bt[n,k] + bias[n] ), 128x128 block tile,
// double-buffered LDS with async global->LDS staging, 1 barrier / K-step.
// 256 threads = 8 waves; wave w -> row-tiles {2*(w&3)+rt}, col-tiles {4*(w>>2)+ct}
template<bool OUT_F32, bool ELU>
__global__ __launch_bounds__(256) void k_gemm(
    const _Float16* __restrict__ Ah, const _Float16* __restrict__ Bt,
    const float* __restrict__ bias,
    _Float16* __restrict__ Oh, float* __restrict__ Of, int K) {
  __shared__ alignas(16) _Float16 As[2][128 * 40];
  __shared__ alignas(16) _Float16 Bs[2][128 * 40];
  const int t    = threadIdx.x;
  const int r    = t >> 1, half = t & 1;       // staging: 2 threads / row
  const int wave = t >> 5, lane = t & 31, hs = lane >> 4, lr = lane & 15;
  const int rt0  = (wave & 3) << 1;            // 2 row tiles
  const int ct0  = (wave >> 2) << 2;           // 4 col tiles
  const size_t m0 = (size_t)blockIdx.y * 128;
  const size_t n0 = (size_t)blockIdx.x * 128;
  const _Float16* Arow = Ah + (m0 + r) * (size_t)K + half * 16;
  const _Float16* Brow = Bt + (n0 + r) * (size_t)K + half * 16;
  const int lofs = r * 40 + half * 16;
  v8f acc[2][4] = {};

  // prologue: stage K-chunk 0 into buffer 0
  g2l_b128(Arow,     &As[0][lofs]);
  g2l_b128(Arow + 8, &As[0][lofs + 8]);
  g2l_b128(Brow,     &Bs[0][lofs]);
  g2l_b128(Brow + 8, &Bs[0][lofs + 8]);
  async_wait();
  __syncthreads();

  const int nit = K >> 5;
  for (int it = 0; it < nit; ++it) {
    const int p = it & 1;
    // stage next chunk into the other buffer (overlaps with WMMA below)
    if (it + 1 < nit) {
      const int kb = (it + 1) << 5;
      g2l_b128(Arow + kb,     &As[1 - p][lofs]);
      g2l_b128(Arow + kb + 8, &As[1 - p][lofs + 8]);
      g2l_b128(Brow + kb,     &Bs[1 - p][lofs]);
      g2l_b128(Brow + kb + 8, &Bs[1 - p][lofs + 8]);
    }
    // ---- WMMA: 2x4 tiles per wave
    v16h af[2];
    for (int rt = 0; rt < 2; ++rt) {
      const int ar = (rt0 + rt) * 16 + lr;
      af[rt] = cat16(*(const v8h*)&As[p][ar * 40 + hs * 16],
                     *(const v8h*)&As[p][ar * 40 + hs * 16 + 8]);
    }
    for (int ct = 0; ct < 4; ++ct) {
      const int bc = (ct0 + ct) * 16 + lr;
      v16h bf = cat16(*(const v8h*)&Bs[p][bc * 40 + hs * 16],
                      *(const v8h*)&Bs[p][bc * 40 + hs * 16 + 8]);
      acc[0][ct] = __builtin_amdgcn_wmma_f32_16x16x32_f16(false, af[0], false, bf,
                                                          (short)0, acc[0][ct], false, false);
      acc[1][ct] = __builtin_amdgcn_wmma_f32_16x16x32_f16(false, af[1], false, bf,
                                                          (short)0, acc[1][ct], false, false);
    }
    async_wait();
    __syncthreads();
  }

  // ---- epilogue: bias (+ elu+1), store f16 or f32
  for (int rt = 0; rt < 2; ++rt) {
    for (int ct = 0; ct < 4; ++ct) {
      const size_t col = n0 + (size_t)(ct0 + ct) * 16 + lr;
      const float bc = bias[col];
      for (int i = 0; i < 8; ++i) {
        const size_t m = m0 + (size_t)(rt0 + rt) * 16 + i + 8 * hs;
        float val = acc[rt][ct][i] + bc;
        if (ELU) val = (val > 0.f) ? (val + 1.f) : __expf(val);
        if (OUT_F32) Of[m * CH + col] = val;
        else         Oh[m * CH + col] = (_Float16)val;
      }
    }
  }
}

// ------------------------------------------------ context = k^T v  (+ k_sum)
// grid: (8 n-splits, 64 bh); block 256. K dim = n (512 per block).
__global__ void k_context(const _Float16* __restrict__ kh, const _Float16* __restrict__ vh,
                          float* __restrict__ ctx, float* __restrict__ ksum) {
  __shared__ alignas(16) _Float16 As[64 * 40];   // [d][n_local]
  __shared__ alignas(16) _Float16 Bs[64 * 40];   // [e][n_local]
  __shared__ float ksacc[64];
  const int t  = threadIdx.x;
  const int bh = blockIdx.y;                     // 0..63
  const int b  = bh >> 3, h = bh & 7;
  const int n0 = blockIdx.x * 512;
  const int nl = t >> 3;                         // 0..31
  const int d0 = (t & 7) * 8;
  const int wave = t >> 5, lane = t & 31, hs = lane >> 4, lr = lane & 15;
  const int tm = wave & 3, tn0 = (wave >> 2) << 1;
  if (t < 64) ksacc[t] = 0.f;
  __syncthreads();
  v8f acc0 = {}; v8f acc1 = {};
  float ksp[8] = {0.f, 0.f, 0.f, 0.f, 0.f, 0.f, 0.f, 0.f};
  const size_t base = (size_t)b * NTOK * CH + (size_t)h * HD;

  for (int kb = 0; kb < 512; kb += 32) {
    const size_t n = (size_t)n0 + kb + nl;
    v8h kv = *(const v8h*)(kh + base + n * CH + d0);
    v8h vv = *(const v8h*)(vh + base + n * CH + d0);
    for (int i = 0; i < 8; ++i) {               // transpose scatter into LDS
      As[(d0 + i) * 40 + nl] = kv[i];
      Bs[(d0 + i) * 40 + nl] = vv[i];
      ksp[i] += (float)kv[i];
    }
    __syncthreads();
    {
      const int ar = tm * 16 + lr;
      v8h a0 = *(const v8h*)&As[ar * 40 + hs * 16];
      v8h a1 = *(const v8h*)&As[ar * 40 + hs * 16 + 8];
      v16h af = cat16(a0, a1);
      int bc = tn0 * 16 + lr;
      v8h b0 = *(const v8h*)&Bs[bc * 40 + hs * 16];
      v8h b1 = *(const v8h*)&Bs[bc * 40 + hs * 16 + 8];
      acc0 = __builtin_amdgcn_wmma_f32_16x16x32_f16(false, af, false, cat16(b0, b1),
                                                    (short)0, acc0, false, false);
      bc += 16;
      b0 = *(const v8h*)&Bs[bc * 40 + hs * 16];
      b1 = *(const v8h*)&Bs[bc * 40 + hs * 16 + 8];
      acc1 = __builtin_amdgcn_wmma_f32_16x16x32_f16(false, af, false, cat16(b0, b1),
                                                    (short)0, acc1, false, false);
    }
    __syncthreads();
  }

  // k_sum reduction
  for (int i = 0; i < 8; ++i) atomicAdd(&ksacc[d0 + i], ksp[i]);
  __syncthreads();
  if (t < 64) atomicAdd(&ksum[(size_t)bh * 64 + t], ksacc[t]);

  // context accumulation across n-splits
  for (int j = 0; j < 2; ++j) {
    v8f acc = j ? acc1 : acc0;
    const int e = (tn0 + j) * 16 + lr;
    for (int i = 0; i < 8; ++i) {
      const int d = tm * 16 + i + 8 * hs;
      atomicAdd(&ctx[(size_t)bh * 4096 + (size_t)d * 64 + e], acc[i]);
    }
  }
}

// ------------------------------- num = q @ ctx ; attn = num / (q.k_sum + eps)
// grid: (64 n-tiles, 64 bh); block 256.
__global__ void k_attn(const _Float16* __restrict__ qh, const float* __restrict__ ctx,
                       const float* __restrict__ ksum, _Float16* __restrict__ attn) {
  __shared__ alignas(16) _Float16 As[64 * 72];   // q tile [row][d]  (K=64)
  __shared__ alignas(16) _Float16 Bs[64 * 72];   // ctx^T  [e][d]
  __shared__ float den[64];
  __shared__ float dpart[256];
  const int t  = threadIdx.x;
  const int bh = blockIdx.y;
  const int b  = bh >> 3, h = bh & 7;
  const int n0 = blockIdx.x * 64;
  const size_t qbase = ((size_t)b * NTOK + n0) * CH + (size_t)h * HD;

  // stage q tile (async straight copy)
  {
    const int r = t >> 2, seg = t & 3;
    for (int s = seg; s < 8; s += 4)
      g2l_b128(qh + qbase + (size_t)r * CH + s * 8, &As[r * 72 + s * 8]);
  }
  // denominator partials: q[row,:] . ksum
  {
    const int row = t & 63, qt = t >> 6;
    float s = 0.f;
    for (int d = qt * 16; d < qt * 16 + 16; ++d)
      s += (float)qh[qbase + (size_t)row * CH + d] * ksum[(size_t)bh * 64 + d];
    dpart[t] = s;
  }
  // stage ctx transposed (f32 -> f16)
  {
    const int d = t >> 2, e0 = (t & 3) * 16;
    const float* cp = ctx + (size_t)bh * 4096 + (size_t)d * 64 + e0;
    for (int i = 0; i < 16; ++i) Bs[(e0 + i) * 72 + d] = (_Float16)cp[i];
  }
  async_wait();
  __syncthreads();
  if (t < 64) den[t] = dpart[t] + dpart[t + 64] + dpart[t + 128] + dpart[t + 192] + 1e-6f;
  __syncthreads();

  const int wave = t >> 5, lane = t & 31, hs = lane >> 4, lr = lane & 15;
  const int tm = wave & 3, tn0 = (wave >> 2) << 1;
  v8f acc0 = {}; v8f acc1 = {};
  for (int kk = 0; kk < 64; kk += 32) {
    const int ar = tm * 16 + lr;
    v8h a0 = *(const v8h*)&As[ar * 72 + kk + hs * 16];
    v8h a1 = *(const v8h*)&As[ar * 72 + kk + hs * 16 + 8];
    v16h af = cat16(a0, a1);
    int bc = tn0 * 16 + lr;
    v8h b0 = *(const v8h*)&Bs[bc * 72 + kk + hs * 16];
    v8h b1 = *(const v8h*)&Bs[bc * 72 + kk + hs * 16 + 8];
    acc0 = __builtin_amdgcn_wmma_f32_16x16x32_f16(false, af, false, cat16(b0, b1),
                                                  (short)0, acc0, false, false);
    bc += 16;
    b0 = *(const v8h*)&Bs[bc * 72 + kk + hs * 16];
    b1 = *(const v8h*)&Bs[bc * 72 + kk + hs * 16 + 8];
    acc1 = __builtin_amdgcn_wmma_f32_16x16x32_f16(false, af, false, cat16(b0, b1),
                                                  (short)0, acc1, false, false);
  }

  for (int j = 0; j < 2; ++j) {
    v8f acc = j ? acc1 : acc0;
    const int e = (tn0 + j) * 16 + lr;
    for (int i = 0; i < 8; ++i) {
      const int rl = tm * 16 + i + 8 * hs;
      float val = acc[i] / den[rl];
      attn[((size_t)b * NTOK + n0 + rl) * CH + (size_t)h * HD + e] = (_Float16)val;
    }
  }
}

// ---------------------------------------------------------------- launcher
extern "C" void kernel_launch(void* const* d_in, const int* in_sizes, int n_in,
                              void* d_out, int out_size, void* d_ws, size_t ws_size,
                              hipStream_t stream) {
  const float* x  = (const float*)d_in[0];
  const float* Wq = (const float*)d_in[1];
  const float* bq = (const float*)d_in[2];
  const float* Wk = (const float*)d_in[3];
  const float* bk = (const float*)d_in[4];
  const float* Wv = (const float*)d_in[5];
  const float* bv = (const float*)d_in[6];
  const float* Wo = (const float*)d_in[7];
  const float* bo = (const float*)d_in[8];
  float* out = (float*)d_out;

  const size_t WELEMS = (size_t)CH * CH;             // 262144
  const size_t QELEMS = (size_t)MROWS * CH;          // 16777216
  _Float16* Wqt  = (_Float16*)d_ws;
  _Float16* Wkt  = Wqt + WELEMS;
  _Float16* Wvt  = Wkt + WELEMS;
  _Float16* Wot  = Wvt + WELEMS;
  _Float16* xh   = Wot + WELEMS;                     // x in f16
  _Float16* q16  = xh  + QELEMS;
  _Float16* k16  = q16 + QELEMS;
  _Float16* v16  = k16 + QELEMS;
  _Float16* attn = xh;                               // alias: xh dead after v-GEMM
  float* ctx  = (float*)(v16 + QELEMS);              // 64*64*64 = 262144 f32
  float* ksum = ctx + (size_t)64 * 4096;             // 64*64 f32 (contiguous)

  // zero ctx + ksum (266240 floats)
  k_zero<<<1040, 256, 0, stream>>>(ctx, 64 * 4096 + 64 * 64);
  // input/weight conversion
  k_xconv<<<8192, 256, 0, stream>>>(x, xh);
  k_wconv<<<1024, 256, 0, stream>>>(Wq, Wqt);
  k_wconv<<<1024, 256, 0, stream>>>(Wk, Wkt);
  k_wconv<<<1024, 256, 0, stream>>>(Wv, Wvt);
  k_wconv<<<1024, 256, 0, stream>>>(Wo, Wot);

  dim3 gproj(CH / 128, MROWS / 128);                 // (4, 256)
  k_gemm<false, true ><<<gproj, 256, 0, stream>>>(xh, Wqt, bq, q16, nullptr, CH);
  k_gemm<false, true ><<<gproj, 256, 0, stream>>>(xh, Wkt, bk, k16, nullptr, CH);
  k_gemm<false, false><<<gproj, 256, 0, stream>>>(xh, Wvt, bv, v16, nullptr, CH);

  k_context<<<dim3(8, 64), 256, 0, stream>>>(k16, v16, ctx, ksum);
  k_attn<<<dim3(64, 64), 256, 0, stream>>>(q16, ctx, ksum, attn);

  k_gemm<true,  false><<<gproj, 256, 0, stream>>>(attn, Wot, bo, nullptr, out, CH);
}
